// DSQGAttentionN_35356170780874
// MI455X (gfx1250) — matrix operand order repaired and checked by reading, not compiled, not run on hardware
//
#include <hip/hip_runtime.h>
#include <hip/hip_bf16.h>

// ---------------------------------------------------------------------------
// Types for CDNA5 WMMA
// ---------------------------------------------------------------------------
typedef __attribute__((ext_vector_type(16))) __bf16 bf16x16;
typedef __attribute__((ext_vector_type(8)))  float  f32x8;

struct Frag32B { uint4 lo, hi; };   // 32 bytes == one bf16x16 operand

__device__ __forceinline__ __bf16 f2bf(float f) {
    unsigned u = __float_as_uint(f);
    unsigned r = (u + 0x7FFFu + ((u >> 16) & 1u)) >> 16;   // round-to-nearest-even
    unsigned short s = (unsigned short)r;
    return __builtin_bit_cast(__bf16, s);
}

// CDNA5 async copy: global -> LDS, 16B per lane, tracked by ASYNCcnt.
__device__ __forceinline__ void async_ld_b128(unsigned lds_off, const void* gaddr) {
    asm volatile("global_load_async_to_lds_b128 %0, %1, off"
                 :: "v"(lds_off), "v"(gaddr) : "memory");
}
__device__ __forceinline__ void wait_asynccnt0() {
    asm volatile("s_wait_asynccnt 0" ::: "memory");
}

// ---------------------------------------------------------------------------
// Elementwise f32 -> bf16 convert
// ---------------------------------------------------------------------------
__global__ __launch_bounds__(256) void cvt_f32_bf16(const float* __restrict__ in,
                                                    __bf16* __restrict__ out, int count) {
    int i = blockIdx.x * 256 + threadIdx.x;
    if (i < count) out[i] = f2bf(in[i]);
}

// f32 [K,N] row-major -> bf16 [N,K] row-major (transpose at convert time so the
// GEMM's B operand stages into LDS exactly like the A operand: K-contiguous rows)
__global__ __launch_bounds__(256) void cvt_f32_bf16_T(const float* __restrict__ in,
                                                      __bf16* __restrict__ out, int K, int N) {
    int i = blockIdx.x * 256 + threadIdx.x;
    if (i < K * N) {
        int k = i / N, n = i % N;
        out[(size_t)n * K + k] = f2bf(in[i]);
    }
}

// ---------------------------------------------------------------------------
// bf16 WMMA GEMM: C[M,N] = A[M,K] @ Bt[N,K]^T (+bias, + fused epilogue)
//   MODE 0: C (f32) = acc + bias[n]
//   MODE 1: C (bf16) = sigmoid(acc + bias[n]) * extra[m,n]
//
// Block tile 128x256, K-step 32; 256 threads = 8 waves (wave32), wave grid
// 2x4, each wave computes 64x64 via 4x4 tiles of v_wmma_f32_16x16x32_bf16
// (16 WMMA : 16 ds_load_b128 per k-step per wave).
//
// Staging uses GLOBAL_LOAD_ASYNC_TO_LDS_B128 (ASYNCcnt) with double-buffered
// LDS: async fills for tile k+1 overlap WMMA math on tile k; one barrier/iter.
// ---------------------------------------------------------------------------
#define BM 128
#define BN 256
#define BK 32
#define LDSB 80                         // row pitch: 64B data + 16B pad (16B-aligned)
#define TILE_A_BYTES (BM * LDSB)        // 10240
#define TILE_B_BYTES (BN * LDSB)        // 20480
#define BUF_BYTES (TILE_A_BYTES + TILE_B_BYTES)  // 30720

template <int MODE>
__global__ __launch_bounds__(256) void gemm_bf16_wmma(
        const __bf16* __restrict__ A,    // [M,K]
        const __bf16* __restrict__ Bt,   // [N,K]
        const float*  __restrict__ bias, // [N]
        const float*  __restrict__ extra,// [M,N] (MODE 1 only)
        void*         __restrict__ Cout,
        int M, int N, int K)
{
    __shared__ __align__(16) unsigned char lds[2 * BUF_BYTES];   // 60 KB
    const unsigned ldsBase = (unsigned)(size_t)&lds[0];

    const int tid   = threadIdx.x;
    const int lane  = tid & 31;
    const int wv    = tid >> 5;      // 0..7
    const int waveM = wv >> 2;       // 0..1 -> 64-row slab
    const int waveN = wv & 3;        // 0..3 -> 64-col slab
    const int blockM = blockIdx.y * BM;
    const int blockN = blockIdx.x * BN;

    const int laneLo = lane & 15;
    const int laneHi = lane >> 4;

    f32x8 acc[4][4] = {};

    // Stage one 128x32 A tile + 256x32 B tile into LDS buffer `buf` for k0.
    // 16B chunks: A = 512 chunks, B = 1024 chunks; 6 async b128 per thread.
    auto stage = [&](int buf, int k0) {
        const unsigned aoff = ldsBase + buf * BUF_BYTES;
        const unsigned boff = aoff + TILE_A_BYTES;
        #pragma unroll
        for (int i = 0; i < 2; ++i) {
            const int idx = tid + 256 * i;           // 0..511
            const int row = idx >> 2, q = idx & 3;   // 4 x 16B chunks per row
            async_ld_b128(aoff + row * LDSB + q * 16,
                          A + (size_t)(blockM + row) * K + k0 + q * 8);
        }
        #pragma unroll
        for (int i = 0; i < 4; ++i) {
            const int idx = tid + 256 * i;           // 0..1023
            const int row = idx >> 2, q = idx & 3;
            async_ld_b128(boff + row * LDSB + q * 16,
                          Bt + (size_t)(blockN + row) * K + k0 + q * 8);
        }
    };

    stage(0, 0);
    const int nIter = K / BK;

    for (int it = 0; it < nIter; ++it) {
        const int buf = it & 1;
        wait_asynccnt0();          // my async fills of `buf` have landed in LDS
        __syncthreads();           // everyone's fills landed; prior reads of other buf done
        if (it + 1 < nIter) stage(1 - buf, (it + 1) * BK);  // overlap with WMMA below

        const unsigned char* curA = lds + buf * BUF_BYTES;
        const unsigned char* curB = curA + TILE_A_BYTES;

        // A 16x32 bf16 fragment: lane(M)=laneLo; K halves {0..7,16..23}/{8..15,24..31}
        bf16x16 av[4];
        #pragma unroll
        for (int tm = 0; tm < 4; ++tm) {
            const unsigned char* p = curA + (waveM * 64 + tm * 16 + laneLo) * LDSB + laneHi * 16;
            Frag32B f; f.lo = *(const uint4*)p; f.hi = *(const uint4*)(p + 32);
            av[tm] = __builtin_bit_cast(bf16x16, f);
        }
        // B 32x16 bf16 fragment: lane(N)=laneLo; K halves {0..7,16..23}/{8..15,24..31}
        bf16x16 bv[4];
        #pragma unroll
        for (int tn = 0; tn < 4; ++tn) {
            const unsigned char* p = curB + (waveN * 64 + tn * 16 + laneLo) * LDSB + laneHi * 16;
            Frag32B f; f.lo = *(const uint4*)p; f.hi = *(const uint4*)(p + 32);
            bv[tn] = __builtin_bit_cast(bf16x16, f);
        }

        #pragma unroll
        for (int tm = 0; tm < 4; ++tm)
            #pragma unroll
            for (int tn = 0; tn < 4; ++tn)
                acc[tm][tn] = __builtin_amdgcn_wmma_f32_16x16x32_bf16(
                    false, av[tm], false, bv[tn], (short)0, acc[tm][tn], false, false);
    }

    // ---- epilogue ----
    #pragma unroll
    for (int tm = 0; tm < 4; ++tm) {
        #pragma unroll
        for (int tn = 0; tn < 4; ++tn) {
            const int mbase = blockM + waveM * 64 + tm * 16 + laneHi * 8;
            const int ncol  = blockN + waveN * 64 + tn * 16 + laneLo;
            const float bn  = bias[ncol];
            #pragma unroll
            for (int r = 0; r < 8; ++r) {
                const size_t idx = (size_t)(mbase + r) * N + ncol;
                float v = acc[tm][tn][r] + bn;
                if (MODE == 0) {
                    ((float*)Cout)[idx] = v;
                } else {
                    float g = 1.0f / (1.0f + __expf(-v));
                    ((__bf16*)Cout)[idx] = f2bf(g * extra[idx]);
                }
            }
        }
    }
}

// ---------------------------------------------------------------------------
// 44-tap causal relative-offset attention.
// One wave per (b,h,n): 32 lanes x float2 = HD=64. Coalesced 256B k/v row
// reads (L2-resident), shfl_xor reduction for the 44 q.k dots, register
// softmax, weighted v accumulation. Writes attn flat [B*N, D] (f32).
// ---------------------------------------------------------------------------
__global__ __launch_bounds__(256) void attn_tap_kernel(
        const float* __restrict__ qkv,      // [B*N, 3*1024]
        const float* __restrict__ pos_bias, // [44,16]
        float* __restrict__ attn)           // [B*N, 1024]
{
    constexpr int NOFF = 44;
    constexpr int offs[NOFF] = {
        0,1,2,3,4,5,6,7,8,9,10,11,12,13,14,15,16,17,18,19,20,21,22,23,24,25,
        26,27,28,29,30,31,32,48,64,96,128,192,256,384,512,768,1024,1536};

    const int lane = threadIdx.x & 31;
    const int wid  = blockIdx.x * 8 + (threadIdx.x >> 5);
    const int b  = wid >> 15;        // H*N = 32768
    const int hn = wid & 32767;
    const int h  = hn >> 11;         // N = 2048
    const int n  = hn & 2047;

    const int dco = h * 64 + lane * 2;
    const float2 q2 = *(const float2*)(qkv + (size_t)(b * 2048 + n) * 3072 + dco);

    float s[NOFF];
    #pragma unroll
    for (int o = 0; o < NOFF; ++o) {
        const int m = n - offs[o];
        float sc = -1e30f;
        if (m >= 0) {
            const float2 k2 = *(const float2*)(qkv + (size_t)(b * 2048 + m) * 3072 + 1024 + dco);
            float p = q2.x * k2.x + q2.y * k2.y;
            p += __shfl_xor(p, 16, 32);
            p += __shfl_xor(p,  8, 32);
            p += __shfl_xor(p,  4, 32);
            p += __shfl_xor(p,  2, 32);
            p += __shfl_xor(p,  1, 32);
            sc = p * 0.125f + pos_bias[o * 16 + h];   // scale = 64^-0.5
        }
        s[o] = sc;
    }

    float mx = s[0];
    #pragma unroll
    for (int o = 1; o < NOFF; ++o) mx = fmaxf(mx, s[o]);
    float sum = 0.f;
    #pragma unroll
    for (int o = 0; o < NOFF; ++o) { float e = __expf(s[o] - mx); s[o] = e; sum += e; }
    const float inv = 1.f / sum;

    float2 a2; a2.x = 0.f; a2.y = 0.f;
    #pragma unroll
    for (int o = 0; o < NOFF; ++o) {
        const int m = n - offs[o];
        if (m >= 0) {
            const float2 v2 = *(const float2*)(qkv + (size_t)(b * 2048 + m) * 3072 + 2048 + dco);
            const float w = s[o] * inv;
            a2.x += w * v2.x;
            a2.y += w * v2.y;
        }
    }
    *(float2*)(attn + (size_t)(b * 2048 + n) * 1024 + dco) = a2;
}

// ---------------------------------------------------------------------------
// Launch
// ---------------------------------------------------------------------------
extern "C" void kernel_launch(void* const* d_in, const int* in_sizes, int n_in,
                              void* d_out, int out_size, void* d_ws, size_t ws_size,
                              hipStream_t stream) {
    (void)in_sizes; (void)n_in; (void)out_size; (void)ws_size;
    const float* x     = (const float*)d_in[0];   // [4096,1024]
    const float* Wqkv  = (const float*)d_in[1];   // [1024,3072]
    const float* bqkv  = (const float*)d_in[2];   // [3072]
    const float* Wout  = (const float*)d_in[3];   // [1024,1024]
    const float* bout  = (const float*)d_in[4];   // [1024]
    const float* Wgate = (const float*)d_in[5];   // [1024,1024]
    const float* bgate = (const float*)d_in[6];   // [1024]
    const float* pbias = (const float*)d_in[7];   // [44,16]
    float* out = (float*)d_out;                   // [4096,1024]

    char* ws = (char*)d_ws;
    __bf16* x_bf   = (__bf16*)(ws);                      // [ 0,  8) MB
    __bf16* wqkvT  = (__bf16*)(ws + ( 8ull << 20));      // [ 8, 14) MB  [3072,1024]
    __bf16* wgateT = (__bf16*)(ws + (14ull << 20));      // [14, 16) MB  [1024,1024]
    __bf16* woutT  = (__bf16*)(ws + (16ull << 20));      // [16, 18) MB  [1024,1024]
    float*  qkv    = (float*) (ws + (18ull << 20));      // [18, 66) MB  [4096,3072]
    float*  attn   = (float*) (ws + (66ull << 20));      // [66, 82) MB  [4096,1024]
    __bf16* y_bf   = (__bf16*)(ws + (82ull << 20));      // [82, 90) MB  [4096,1024]

    const int M = 4096, D = 1024;

    // 1) precision casts (weights transposed to [N,K] for the WMMA B operand)
    cvt_f32_bf16  <<<(M * D + 255) / 256,      256, 0, stream>>>(x, x_bf, M * D);
    cvt_f32_bf16_T<<<(D * 3 * D + 255) / 256,  256, 0, stream>>>(Wqkv,  wqkvT,  D, 3 * D);
    cvt_f32_bf16_T<<<(D * D + 255) / 256,      256, 0, stream>>>(Wgate, wgateT, D, D);
    cvt_f32_bf16_T<<<(D * D + 255) / 256,      256, 0, stream>>>(Wout,  woutT,  D, D);

    // 2) qkv = x @ Wqkv + bqkv                 (WMMA, f32 out)
    gemm_bf16_wmma<0><<<dim3(3 * D / BN, M / BM), 256, 0, stream>>>(
        x_bf, wqkvT, bqkv, nullptr, qkv, M, 3 * D, D);

    // 3) 44-tap attention -> attn flat [M, D]
    attn_tap_kernel<<<(2 * 16 * 2048) / 8, 256, 0, stream>>>(qkv, pbias, attn);

    // 4) y = sigmoid(x @ Wgate + bgate) * attn (WMMA, bf16 out, fused gate)
    gemm_bf16_wmma<1><<<dim3(D / BN, M / BM), 256, 0, stream>>>(
        x_bf, wgateT, bgate, attn, y_bf, M, D, D);

    // 5) out = y @ Wout + bout                 (WMMA, f32 out)
    gemm_bf16_wmma<0><<<dim3(D / BN, M / BM), 256, 0, stream>>>(
        y_bf, woutT, bout, nullptr, out, M, D, D);
}
